// Head_17575006175767
// MI455X (gfx1250) — matrix operand order repaired and checked by reading, not compiled, not run on hardware
//
#include <hip/hip_runtime.h>
#include <hip/hip_bf16.h>
#include <math.h>

// Shapes (fixed by the reference)
#define HEAD  64
#define NEMBD 1024
#define TSEQ  2048
#define NB    8

typedef __attribute__((ext_vector_type(16))) __bf16 v16bf;
typedef __attribute__((ext_vector_type(8)))  __bf16 v8bf;
typedef __attribute__((ext_vector_type(8)))  float  v8f;
typedef __attribute__((ext_vector_type(4)))  float  v4f;

// ---------------------------------------------------------------------------
// Kernel 0: one-time fp32 -> bf16 conversion of the three weight matrices,
// stored concatenated [3][64][1024] bf16.  768 KB read / 384 KB write.
// ---------------------------------------------------------------------------
__global__ __launch_bounds__(256) void wconv_kernel(
    const float* __restrict__ Wk,
    const float* __restrict__ Wq,
    const float* __restrict__ Wv,
    __bf16* __restrict__ Wb)             // [3*64*1024]
{
    const int idx = (blockIdx.x * 256 + threadIdx.x) * 4;   // float4 granularity
    const int per = HEAD * NEMBD;                           // 65536
    const int m   = idx / per;
    const int off = idx % per;
    const float* src = (m == 0) ? Wk : (m == 1) ? Wq : Wv;
    v4f f = *(const v4f*)(src + off);
    v4f* dummy; (void)dummy;
    __bf16* dst = Wb + (size_t)m * per + off;
    dst[0] = (__bf16)f.x; dst[1] = (__bf16)f.y;
    dst[2] = (__bf16)f.z; dst[3] = (__bf16)f.w;
}

// ---------------------------------------------------------------------------
// Kernel 1: q/k/v projections.  One wave (32 threads) computes a 16-token tile
// of all three 64-wide projections with v_wmma_f32_16x16x32_bf16.
// B fragments now come straight from bf16 W (2x global_load_b128 each).
//   k,q stored row-major bf16 [B*T, 64];  v stored transposed bf16 [B, 64, T]
// ---------------------------------------------------------------------------
__global__ __launch_bounds__(32) void proj_kernel(
    const float*  __restrict__ x,
    const __bf16* __restrict__ Wb,   // [3][64][1024] bf16
    __bf16* __restrict__ kq,
    __bf16* __restrict__ qq,
    __bf16* __restrict__ vT)
{
    const int lane  = threadIdx.x;
    const int hf    = lane >> 4;     // lane half (0/1)
    const int l16   = lane & 15;
    const int mbase = blockIdx.x * 16;

    v8f acc[3][4] = {};

    const int arow = mbase + l16;    // A-matrix row for this lane
    const int koff = hf * 16;        // A layout: lanes 16-31 hold K+16..31

    // Per-lane W base: B frag col n = l16, K contiguous in the bf16 W row
    const __bf16* wlane = Wb + (size_t)l16 * NEMBD + hf * 16;

    for (int k0 = 0; k0 < NEMBD; k0 += 32) {
        // A fragment (16x32 bf16) from x, converted fp32 -> bf16 on load
        v16bf a;
        {
            const float* ap = x + (size_t)arow * NEMBD + k0 + koff;
            #pragma unroll
            for (int i = 0; i < 8; ++i) a[i]     = (__bf16)ap[i];       // K = koff..koff+7
            #pragma unroll
            for (int i = 0; i < 8; ++i) a[8 + i] = (__bf16)ap[8 + i];   // K = koff+8..koff+15
        }
        #pragma unroll
        for (int m = 0; m < 3; ++m) {
            #pragma unroll
            for (int nt = 0; nt < 4; ++nt) {
                const __bf16* bp = wlane
                    + (size_t)m * (HEAD * NEMBD)
                    + (size_t)(nt * 16) * NEMBD + k0;
                v16bf bf = *(const v16bf*)bp;   // 32B: K = k0 + hf*16 + 0..15
                acc[m][nt] = __builtin_amdgcn_wmma_f32_16x16x32_bf16(
                    false, a, false, bf, (short)0, acc[m][nt], false, false);
            }
        }
    }

    // C layout: element i -> row (i + 8*hf), col l16
    const int b  = mbase / TSEQ;
    const int t0 = mbase % TSEQ;
    #pragma unroll
    for (int nt = 0; nt < 4; ++nt) {
        #pragma unroll
        for (int i = 0; i < 8; ++i) {
            const int r = hf * 8 + i;
            const int h = nt * 16 + l16;
            const size_t row = (size_t)(mbase + r);
            kq[row * HEAD + h] = (__bf16)acc[0][nt][i];
            qq[row * HEAD + h] = (__bf16)acc[1][nt][i];
            vT[((size_t)b * HEAD + h) * TSEQ + (t0 + r)] = (__bf16)acc[2][nt][i];
        }
    }
}

// ---------------------------------------------------------------------------
// Kernel 2: causal flash attention.  One wave per 16-query tile; keys streamed
// 32 at a time.  S via 4 WMMAs (K=32 over head dim), online softmax with
// half-wave shfl_xor reductions, P transposed through LDS, O += P*V.
// ---------------------------------------------------------------------------
__global__ __launch_bounds__(32) void attn_kernel(
    const __bf16* __restrict__ kq,   // [B*T, 64]
    const __bf16* __restrict__ qq,   // [B*T, 64]
    const __bf16* __restrict__ vT,   // [B, 64, T]
    float* __restrict__ out)         // [B*T, 64]
{
    __shared__ __align__(32) __bf16 pTile[16 * 32];

    const int lane  = threadIdx.x;
    const int hf    = lane >> 4;
    const int l16   = lane & 15;
    const int qbase = blockIdx.x * 16;
    const int b     = blockIdx.y;
    const size_t rowbase = (size_t)b * TSEQ;

    // Q A-fragments for head-dim K-blocks c0 = 0 and 32
    v16bf aq[2];
    {
        const __bf16* qp = qq + (rowbase + qbase + l16) * HEAD + hf * 16;
        #pragma unroll
        for (int cc = 0; cc < 2; ++cc) {
            v8bf lo = *(const v8bf*)(qp + cc * 32);
            v8bf hi = *(const v8bf*)(qp + cc * 32 + 8);
            #pragma unroll
            for (int i = 0; i < 8; ++i) { aq[cc][i] = lo[i]; aq[cc][8 + i] = hi[i]; }
        }
    }

    float mrow[8], lrow[8];
    #pragma unroll
    for (int i = 0; i < 8; ++i) { mrow[i] = -INFINITY; lrow[i] = 0.f; }
    v8f o[4] = {};

    const float scale = 0.03125f;  // 1024^-0.5 (reference scales by n_embd)

    for (int kb = 0; kb < qbase + 16; kb += 32) {
        // ---- S = Q K^T for 32 keys (two 16-key C tiles) ----
        v8f s[2];
        #pragma unroll
        for (int t = 0; t < 2; ++t) {
            const __bf16* kp = kq + (rowbase + kb + t * 16 + l16) * HEAD + hf * 16;
            v16bf b0 = *(const v16bf*)(kp);        // features 0..31
            v16bf b1 = *(const v16bf*)(kp + 32);   // features 32..63
            v8f z = {};
            z = __builtin_amdgcn_wmma_f32_16x16x32_bf16(false, aq[0], false, b0, (short)0, z, false, false);
            z = __builtin_amdgcn_wmma_f32_16x16x32_bf16(false, aq[1], false, b1, (short)0, z, false, false);
            s[t] = z;
        }

        // ---- scale + causal mask ----
        const bool needMask = (kb + 31 > qbase);
        #pragma unroll
        for (int t = 0; t < 2; ++t) {
            #pragma unroll
            for (int i = 0; i < 8; ++i) {
                float v = s[t][i] * scale;
                if (needMask) {
                    const int r = hf * 8 + i;
                    const int c = t * 16 + l16;
                    if (kb + c > qbase + r) v = -INFINITY;
                }
                s[t][i] = v;
            }
        }

        // ---- online softmax (rows live in (element, lane-half); xor 1/2/4/8
        //      reductions stay inside each 16-lane half) ----
        #pragma unroll
        for (int i = 0; i < 8; ++i) {
            float mx = fmaxf(s[0][i], s[1][i]);
            #pragma unroll
            for (int off = 1; off < 16; off <<= 1)
                mx = fmaxf(mx, __shfl_xor(mx, off, 32));
            const float mnew  = fmaxf(mrow[i], mx);
            const float alpha = __expf(mrow[i] - mnew);
            const float p0 = __expf(s[0][i] - mnew);
            const float p1 = __expf(s[1][i] - mnew);
            float rs = p0 + p1;
            #pragma unroll
            for (int off = 1; off < 16; off <<= 1)
                rs += __shfl_xor(rs, off, 32);
            lrow[i] = lrow[i] * alpha + rs;
            mrow[i] = mnew;
            s[0][i] = p0; s[1][i] = p1;
            #pragma unroll
            for (int h = 0; h < 4; ++h) o[h][i] *= alpha;
        }

        // ---- transpose P (C layout) -> A layout via LDS ----
        __syncthreads();  // single-wave WG: S_NOP + waitcnt
        #pragma unroll
        for (int t = 0; t < 2; ++t) {
            #pragma unroll
            for (int i = 0; i < 8; ++i) {
                const int r = hf * 8 + i;
                const int c = t * 16 + l16;
                pTile[r * 32 + c] = (__bf16)s[t][i];
            }
        }
        __syncthreads();
        v16bf ap;
        {
            const __bf16* pp = pTile + l16 * 32 + hf * 16;
            v8bf lo = *(const v8bf*)(pp);
            v8bf hi = *(const v8bf*)(pp + 8);
            #pragma unroll
            for (int i = 0; i < 8; ++i) { ap[i] = lo[i]; ap[8 + i] = hi[i]; }
        }

        // ---- O += P * V  (B fragments contiguous thanks to transposed V) ----
        const __bf16* vp = vT + ((size_t)b * HEAD + l16) * TSEQ + kb + hf * 16;
        #pragma unroll
        for (int h = 0; h < 4; ++h) {
            v16bf bv = *(const v16bf*)(vp + (size_t)h * 16 * TSEQ);
            o[h] = __builtin_amdgcn_wmma_f32_16x16x32_bf16(
                false, ap, false, bv, (short)0, o[h], false, false);
        }
    }

    // ---- normalize and store fp32 output ----
    #pragma unroll
    for (int i = 0; i < 8; ++i) {
        const float inv = 1.f / lrow[i];
        const int r = hf * 8 + i;
        float* op = out + (rowbase + qbase + r) * HEAD + l16;
        #pragma unroll
        for (int h = 0; h < 4; ++h)
            op[h * 16] = o[h][i] * inv;
    }
}

extern "C" void kernel_launch(void* const* d_in, const int* in_sizes, int n_in,
                              void* d_out, int out_size, void* d_ws, size_t ws_size,
                              hipStream_t stream) {
    const float* x  = (const float*)d_in[0];
    const float* Wk = (const float*)d_in[1];
    const float* Wq = (const float*)d_in[2];
    const float* Wv = (const float*)d_in[3];
    float* out = (float*)d_out;

    const size_t NTOK = (size_t)NB * TSEQ;
    __bf16* kq = (__bf16*)d_ws;            // [B*T,64] bf16  (2 MB)
    __bf16* qq = kq + NTOK * HEAD;         // [B*T,64] bf16  (2 MB)
    __bf16* vT = qq + NTOK * HEAD;         // [B,64,T] bf16  (2 MB)
    __bf16* Wb = vT + NTOK * HEAD;         // [3,64,1024] bf16 (384 KB)

    // 3*64*1024 floats, 4 per thread, 256 threads/block -> 192 blocks
    wconv_kernel<<<dim3(192), 256, 0, stream>>>(Wk, Wq, Wv, Wb);
    proj_kernel<<<dim3((unsigned)(NTOK / 16)), 32, 0, stream>>>(x, Wb, kq, qq, vT);
    attn_kernel<<<dim3(TSEQ / 16, NB), 32, 0, stream>>>(kq, qq, vT, out);
}
